// TransformerEncoderLayer_37890201485508
// MI455X (gfx1250) — compile-verified, matrix operand order checked by reading, not run on hardware
//
#include <hip/hip_runtime.h>
#include <hip/hip_bf16.h>
#include <math.h>

// ---------------------------------------------------------------------------
// Types / WMMA helpers (CDNA5 / gfx1250, wave32)
// ---------------------------------------------------------------------------
typedef __attribute__((ext_vector_type(16))) __bf16 v16bf;
typedef __attribute__((ext_vector_type(8)))  __bf16 v8bf;
typedef __attribute__((ext_vector_type(8)))  float  v8f;

#define WMMA_BF16(a, b, c) \
  __builtin_amdgcn_wmma_f32_16x16x32_bf16(false, (a), false, (b), (short)0, (c), false, false)

static __device__ __forceinline__ __bf16 f2bf(float f) {
  union { float f; unsigned u; } in; in.f = f;
  unsigned u = in.u;
  u += 0x7fffu + ((u >> 16) & 1u);                       // round-to-nearest-even
  union { unsigned short s; __bf16 b; } o; o.s = (unsigned short)(u >> 16);
  return o.b;
}

static __device__ __forceinline__ float gelu_exact(float v) {
  return 0.5f * v * (1.0f + erff(v * 0.70710678118654752f));
}

union Frag16 { v16bf v; v8bf h[2]; };

// 16-bit A/B fragment: lane (l&15) picks the row/column, (l>>4) selects which
// half of the K pattern {0..7,16..23} vs {8..15,24..31}.  Caller passes the
// row-base pointer already offset by k0 + ((lane>>4)<<3); we grab the two
// contiguous 8-element chunks at +0 and +16.
static __device__ __forceinline__ v16bf load_frag(const __bf16* p) {
  Frag16 f;
  f.h[0] = *(const v8bf*)(p);
  f.h[1] = *(const v8bf*)(p + 16);
  return f.v;
}

static __device__ __forceinline__ v8f vzero8() {
  v8f z = {0.f,0.f,0.f,0.f,0.f,0.f,0.f,0.f};
  return z;
}

// ---------------------------------------------------------------------------
// Problem dims
// ---------------------------------------------------------------------------
#define Bn   64
#define Nn   256
#define Dn   256
#define Hn   8
#define En   12
#define Fn   1024
#define DH   32
#define Mrows (Bn*Nn)          // 16384
static __device__ const float kScale = 0.17677669529663687f;   // 32^-0.5

// ---------------------------------------------------------------------------
// fp32 -> bf16 cast (grid-stride)
// ---------------------------------------------------------------------------
__global__ void cast_bf16_kernel(const float* __restrict__ s, __bf16* __restrict__ d, int n) {
  for (int i = blockIdx.x * blockDim.x + threadIdx.x; i < n; i += gridDim.x * blockDim.x)
    d[i] = f2bf(s[i]);
}

// ---------------------------------------------------------------------------
// Fused QKV projection:  [16384,256] x [768,256]^T -> scatter to
//   Q,K : [B,H,N,32] bf16   (row = token, contiguous Dh)
//   Vt  : [B,H,32,N] bf16   (transposed so AV B-fragments load contiguously)
// grid = (M/16, 768/128), block = 256 (8 waves, wave = 16x16 tile)
// ---------------------------------------------------------------------------
__global__ void qkv_kernel(const __bf16* __restrict__ A, const __bf16* __restrict__ W,
                           const float* __restrict__ bq, const float* __restrict__ bk,
                           const float* __restrict__ bv,
                           __bf16* __restrict__ Q, __bf16* __restrict__ K,
                           __bf16* __restrict__ Vt) {
  const int lane = threadIdx.x & 31, wv = threadIdx.x >> 5;
  const int m0 = blockIdx.x * 16;
  const int n0 = blockIdx.y * 128 + wv * 16;
  const int ksel = (lane >> 4) << 3;
  const __bf16* ap = A + (size_t)(m0 + (lane & 15)) * Dn + ksel;
  const __bf16* wp = W + (size_t)(n0 + (lane & 15)) * Dn + ksel;
  v8f acc = vzero8();
  for (int k0 = 0; k0 < Dn; k0 += 32) {
    v16bf a = load_frag(ap + k0);
    v16bf b = load_frag(wp + k0);
    acc = WMMA_BF16(a, b, acc);
  }
  const int c = n0 + (lane & 15);       // global output column 0..767
  const int which = c >> 8;             // 0=Q 1=K 2=V
  const int d = c & 255, h = d >> 5, dh = d & 31;
  const float bias = (which == 0 ? bq : which == 1 ? bk : bv)[d];
  const int hi8 = (lane >> 4) << 3;
#pragma unroll
  for (int j = 0; j < 8; ++j) {
    const int m = m0 + j + hi8;         // token index 0..16383
    const int b_ = m >> 8, n_ = m & 255;
    const __bf16 val = f2bf(acc[j] + bias);
    const size_t bh = (size_t)(b_ * Hn + h);
    if (which == 0)      Q[(bh * Nn + n_) * DH + dh] = val;
    else if (which == 1) K[(bh * Nn + n_) * DH + dh] = val;
    else                 Vt[(bh * DH + dh) * Nn + n_] = val;
  }
}

// ---------------------------------------------------------------------------
// Flash attention with edge bias.
// grid = B*(N/16) blocks; block = 256 = 8 waves; wave w <-> head w.
// Edge tile edge[b, q0:q0+16, m0:m0+32, 0:12] is staged into LDS with
// GLOBAL_LOAD_ASYNC_TO_LDS_B128 (ASYNCcnt path, no VGPR round-trip) once per
// key-step and shared by all 8 heads -> edge_attr (201 MB) read exactly once.
// ---------------------------------------------------------------------------
__global__ void attn_kernel(const __bf16* __restrict__ Q, const __bf16* __restrict__ Kb,
                            const __bf16* __restrict__ Vt,
                            const float* __restrict__ edge, const float* __restrict__ We,
                            const float* __restrict__ be, const unsigned char* __restrict__ mask,
                            __bf16* __restrict__ ctx) {
  __shared__ __align__(16) float  lds_e[16 * 32 * En];   // 24 KB
  __shared__ __align__(16) __bf16 lds_p[Hn][16 * 32];    // 8 KB, per-wave P staging
  const int lane = threadIdx.x & 31, h = threadIdx.x >> 5;
  const int b = blockIdx.x >> 4;
  const int q0 = (blockIdx.x & 15) << 4;

  float we[En];
#pragma unroll
  for (int e = 0; e < En; ++e) we[e] = We[h * En + e];
  const float beh = be[h];

  const __bf16* Qh = Q  + (size_t)(b * Hn + h) * Nn * DH;
  const __bf16* Kh = Kb + (size_t)(b * Hn + h) * Nn * DH;
  const __bf16* Vh = Vt + (size_t)(b * Hn + h) * DH * Nn;
  const int ksel = (lane >> 4) << 3;
  const int c0 = lane & 15, c1 = c0 + 16;
  const int hi8 = (lane >> 4) << 3;

  const v16bf qf = load_frag(Qh + (q0 + c0) * DH + ksel);   // A-frag, reused all steps
  v8f acc0 = vzero8(), acc1 = vzero8(), lrow = vzero8();
  v8f mrow;
#pragma unroll
  for (int j = 0; j < 8; ++j) mrow[j] = -1e30f;

  for (int m0 = 0; m0 < Nn; m0 += 32) {
    __syncthreads();
    // async DMA of the edge tile: 16 rows x (32 keys * 12 feats) floats.
    // Each row is 384 contiguous floats (96 float4 chunks, 16B aligned).
    {
      const float* basep = edge + ((size_t)(b * Nn + q0) * Nn + m0) * En;
      for (int g = threadIdx.x; g < 16 * 96; g += 256) {
        const int r = g / 96, w_ = g % 96;
        const float* gp = basep + (size_t)r * (Nn * En) + w_ * 4;
        unsigned lp = (unsigned)(uintptr_t)(const void*)&lds_e[g * 4];
        asm volatile("global_load_async_to_lds_b128 %0, %1, off"
                     :: "v"(lp), "v"(gp) : "memory");
      }
      asm volatile("s_wait_asynccnt 0x0" ::: "memory");
    }
    __syncthreads();

    // scores for 32 keys: two QK^T WMMAs (K = Dh = 32)
    v16bf kf0 = load_frag(Kh + (m0 + c0) * DH + ksel);
    v16bf kf1 = load_frag(Kh + (m0 + 16 + c0) * DH + ksel);
    v8f s0 = WMMA_BF16(qf, kf0, vzero8());
    v8f s1 = WMMA_BF16(qf, kf1, vzero8());

    const unsigned char mk0 = mask[b * Nn + m0 + c0];
    const unsigned char mk1 = mask[b * Nn + m0 + c1];
#pragma unroll
    for (int j = 0; j < 8; ++j) {
      const int r = j + hi8;
      float b0 = beh, b1 = beh;
      const float* e0 = &lds_e[(r * 32 + c0) * En];
      const float* e1 = &lds_e[(r * 32 + c1) * En];
#pragma unroll
      for (int e = 0; e < En; ++e) { b0 += e0[e] * we[e]; b1 += e1[e] * we[e]; }
      s0[j] = mk0 ? s0[j] * kScale + b0 : -1e30f;
      s1[j] = mk1 ? s1[j] * kScale + b1 : -1e30f;
    }

    // online softmax: rows live in 16-lane halves (C/D layout), reduce across them
    v8f rm;
#pragma unroll
    for (int j = 0; j < 8; ++j) rm[j] = fmaxf(s0[j], s1[j]);
    for (int off = 8; off; off >>= 1)
#pragma unroll
      for (int j = 0; j < 8; ++j) rm[j] = fmaxf(rm[j], __shfl_xor(rm[j], off, 16));

    v8f mnew, corr, rs;
#pragma unroll
    for (int j = 0; j < 8; ++j) {
      mnew[j] = fmaxf(mrow[j], rm[j]);
      corr[j] = __expf(mrow[j] - mnew[j]);
      s0[j] = __expf(s0[j] - mnew[j]);
      s1[j] = __expf(s1[j] - mnew[j]);
      rs[j] = s0[j] + s1[j];
    }
    for (int off = 8; off; off >>= 1)
#pragma unroll
      for (int j = 0; j < 8; ++j) rs[j] += __shfl_xor(rs[j], off, 16);
#pragma unroll
    for (int j = 0; j < 8; ++j) {
      lrow[j] = lrow[j] * corr[j] + rs[j];
      mrow[j] = mnew[j];
      acc0[j] *= corr[j];
      acc1[j] *= corr[j];
    }

    // restage P (C/D layout) -> A-fragment layout via this wave's LDS region
    __bf16* pp = lds_p[h];
#pragma unroll
    for (int j = 0; j < 8; ++j) {
      const int r = j + hi8;
      pp[r * 32 + c0] = f2bf(s0[j]);
      pp[r * 32 + c1] = f2bf(s1[j]);
    }
    v16bf pf  = load_frag(pp + c0 * 32 + ksel);
    v16bf vf0 = load_frag(Vh + c0 * Nn + m0 + ksel);          // Dh cols 0..15
    v16bf vf1 = load_frag(Vh + (16 + c0) * Nn + m0 + ksel);   // Dh cols 16..31
    acc0 = WMMA_BF16(pf, vf0, acc0);
    acc1 = WMMA_BF16(pf, vf1, acc1);
  }

#pragma unroll
  for (int j = 0; j < 8; ++j) {
    const int r = j + hi8;
    const float inv = 1.0f / lrow[j];
    const size_t row = (size_t)(b * Nn + q0 + r) * Dn + h * DH;
    ctx[row + c0]      = f2bf(acc0[j] * inv);
    ctx[row + 16 + c0] = f2bf(acc1[j] * inv);
  }
}

// ---------------------------------------------------------------------------
// Generic WMMA GEMM: out = A[M,K] x W[N,K]^T + bias, with epilogue.
// EPI 0: fp32 store.  EPI 1: exact GELU -> bf16 store.
// Wave tile = 32x32 (2 A-frags x 2 B-frags -> 4 WMMAs per K-step, so the
// matrix pipe, not VMEM issue, is the limiter).  Block = 8 waves covering
// 32 rows x 256 cols.  grid = (M/32, Nout/256).
// ---------------------------------------------------------------------------
template <int EPI>
__global__ void gemm_bf16_kernel(const __bf16* __restrict__ A, const __bf16* __restrict__ W,
                                 const float* __restrict__ bias,
                                 float* __restrict__ outf, __bf16* __restrict__ outb,
                                 int Nout, int Kdim) {
  const int lane = threadIdx.x & 31, wv = threadIdx.x >> 5;
  const int m0 = blockIdx.x * 32;
  const int n0 = blockIdx.y * 256 + wv * 32;
  const int ksel = (lane >> 4) << 3;
  const __bf16* ap0 = A + (size_t)(m0 + (lane & 15)) * Kdim + ksel;
  const __bf16* ap1 = ap0 + (size_t)16 * Kdim;
  const __bf16* wp0 = W + (size_t)(n0 + (lane & 15)) * Kdim + ksel;
  const __bf16* wp1 = wp0 + (size_t)16 * Kdim;
  v8f acc00 = vzero8(), acc01 = vzero8(), acc10 = vzero8(), acc11 = vzero8();
  for (int k0 = 0; k0 < Kdim; k0 += 32) {
    v16bf a0 = load_frag(ap0 + k0);
    v16bf a1 = load_frag(ap1 + k0);
    v16bf b0 = load_frag(wp0 + k0);
    v16bf b1 = load_frag(wp1 + k0);
    acc00 = WMMA_BF16(a0, b0, acc00);
    acc01 = WMMA_BF16(a0, b1, acc01);
    acc10 = WMMA_BF16(a1, b0, acc10);
    acc11 = WMMA_BF16(a1, b1, acc11);
  }
  const int cA = n0 + (lane & 15), cB = cA + 16;
  const float bv0 = bias[cA], bv1 = bias[cB];
  const int hi8 = (lane >> 4) << 3;
#pragma unroll
  for (int j = 0; j < 8; ++j) {
    const size_t r0 = (size_t)m0 + j + hi8;
    const size_t r1 = r0 + 16;
    float v00 = acc00[j] + bv0, v01 = acc01[j] + bv1;
    float v10 = acc10[j] + bv0, v11 = acc11[j] + bv1;
    if (EPI == 1) {
      outb[r0 * Nout + cA] = f2bf(gelu_exact(v00));
      outb[r0 * Nout + cB] = f2bf(gelu_exact(v01));
      outb[r1 * Nout + cA] = f2bf(gelu_exact(v10));
      outb[r1 * Nout + cB] = f2bf(gelu_exact(v11));
    } else {
      outf[r0 * Nout + cA] = v00;
      outf[r0 * Nout + cB] = v01;
      outf[r1 * Nout + cA] = v10;
      outf[r1 * Nout + cB] = v11;
    }
  }
}

// ---------------------------------------------------------------------------
// Fused residual + LayerNorm over D=256; wave per row, 8 rows per block.
// Optionally also writes a bf16 copy (feeds the next GEMM's A matrix).
// ---------------------------------------------------------------------------
__global__ void add_ln_kernel(const float* __restrict__ a, const float* __restrict__ r,
                              const float* __restrict__ g, const float* __restrict__ beta,
                              float* __restrict__ outf, __bf16* __restrict__ outb) {
  const int lane = threadIdx.x & 31, wv = threadIdx.x >> 5;
  const int row = blockIdx.x * 8 + wv;
  const float* pa = a + (size_t)row * Dn;
  const float* pr = r + (size_t)row * Dn;
  float x[8], s = 0.f, s2 = 0.f;
#pragma unroll
  for (int i = 0; i < 8; ++i) {
    x[i] = pa[lane + 32 * i] + pr[lane + 32 * i];
    s += x[i]; s2 += x[i] * x[i];
  }
  for (int off = 16; off; off >>= 1) { s += __shfl_xor(s, off, 32); s2 += __shfl_xor(s2, off, 32); }
  const float mean = s * (1.0f / Dn);
  const float var = s2 * (1.0f / Dn) - mean * mean;
  const float rstd = rsqrtf(var + 1e-5f);
#pragma unroll
  for (int i = 0; i < 8; ++i) {
    const int cidx = lane + 32 * i;
    const float y = (x[i] - mean) * rstd * g[cidx] + beta[cidx];
    outf[(size_t)row * Dn + cidx] = y;
    if (outb) outb[(size_t)row * Dn + cidx] = f2bf(y);
  }
}

// ---------------------------------------------------------------------------
// Launcher
// ---------------------------------------------------------------------------
extern "C" void kernel_launch(void* const* d_in, const int* in_sizes, int n_in,
                              void* d_out, int out_size, void* d_ws, size_t ws_size,
                              hipStream_t stream) {
  const float* x    = (const float*)d_in[0];
  const unsigned char* mask = (const unsigned char*)d_in[1];
  const float* edge = (const float*)d_in[2];
  const float* Wq = (const float*)d_in[3],  *bq = (const float*)d_in[4];
  const float* Wk = (const float*)d_in[5],  *bk = (const float*)d_in[6];
  const float* Wv = (const float*)d_in[7],  *bv = (const float*)d_in[8];
  const float* Wo = (const float*)d_in[9],  *bo = (const float*)d_in[10];
  const float* We = (const float*)d_in[11], *be = (const float*)d_in[12];
  const float* g1 = (const float*)d_in[13], *beta1 = (const float*)d_in[14];
  const float* g2 = (const float*)d_in[15], *beta2 = (const float*)d_in[16];
  const float* W1 = (const float*)d_in[17], *b1 = (const float*)d_in[18];
  const float* W2 = (const float*)d_in[19], *b2 = (const float*)d_in[20];
  float* out = (float*)d_out;

  // workspace carve-up (256B aligned)
  char* cur = (char*)d_ws;
  auto alloc = [&](size_t bytes) { void* p = cur; cur += (bytes + 255) & ~(size_t)255; return p; };
  __bf16* xb    = (__bf16*)alloc((size_t)Mrows * Dn * 2);
  __bf16* wqkvb = (__bf16*)alloc((size_t)3 * Dn * Dn * 2);
  __bf16* wob   = (__bf16*)alloc((size_t)Dn * Dn * 2);
  __bf16* w1b   = (__bf16*)alloc((size_t)Fn * Dn * 2);
  __bf16* w2b   = (__bf16*)alloc((size_t)Dn * Fn * 2);
  __bf16* Qb    = (__bf16*)alloc((size_t)Bn * Hn * Nn * DH * 2);
  __bf16* Kbuf  = (__bf16*)alloc((size_t)Bn * Hn * Nn * DH * 2);
  __bf16* Vtb   = (__bf16*)alloc((size_t)Bn * Hn * DH * Nn * 2);
  __bf16* ctxb  = (__bf16*)alloc((size_t)Mrows * Dn * 2);
  float*  attnf = (float*) alloc((size_t)Mrows * Dn * 4);
  float*  h1f   = (float*) alloc((size_t)Mrows * Dn * 4);
  __bf16* h1b   = (__bf16*)alloc((size_t)Mrows * Dn * 2);
  __bf16* tb    = (__bf16*)alloc((size_t)Mrows * Fn * 2);
  float*  f2f   = (float*) alloc((size_t)Mrows * Dn * 4);

  auto cast = [&](const float* s, __bf16* d, int n) {
    int grid = (n + 1023) / 1024; if (grid > 2048) grid = 2048;
    hipLaunchKernelGGL(cast_bf16_kernel, dim3(grid), dim3(256), 0, stream, s, d, n);
  };
  cast(x, xb, Mrows * Dn);
  cast(Wq, wqkvb,                Dn * Dn);
  cast(Wk, wqkvb +     Dn * Dn,  Dn * Dn);
  cast(Wv, wqkvb + 2 * Dn * Dn,  Dn * Dn);
  cast(Wo, wob, Dn * Dn);
  cast(W1, w1b, Fn * Dn);
  cast(W2, w2b, Dn * Fn);

  // QKV projection
  hipLaunchKernelGGL(qkv_kernel, dim3(Mrows / 16, (3 * Dn) / 128), dim3(256), 0, stream,
                     xb, wqkvb, bq, bk, bv, Qb, Kbuf, Vtb);

  // flash attention with edge bias (async-LDS edge staging)
  hipLaunchKernelGGL(attn_kernel, dim3(Bn * (Nn / 16)), dim3(256), 0, stream,
                     Qb, Kbuf, Vtb, edge, We, be, mask, ctxb);

  // output projection
  hipLaunchKernelGGL((gemm_bf16_kernel<0>), dim3(Mrows / 32, Dn / 256), dim3(256), 0, stream,
                     ctxb, wob, bo, attnf, (__bf16*)nullptr, Dn, Dn);

  // residual + LN1 (fp32 + bf16 copies)
  hipLaunchKernelGGL(add_ln_kernel, dim3(Mrows / 8), dim3(256), 0, stream,
                     x, attnf, g1, beta1, h1f, h1b);

  // FFN up + exact GELU -> bf16
  hipLaunchKernelGGL((gemm_bf16_kernel<1>), dim3(Mrows / 32, Fn / 256), dim3(256), 0, stream,
                     h1b, w1b, b1, (float*)nullptr, tb, Fn, Dn);

  // FFN down
  hipLaunchKernelGGL((gemm_bf16_kernel<0>), dim3(Mrows / 32, Dn / 256), dim3(256), 0, stream,
                     tb, w2b, b2, f2f, (__bf16*)nullptr, Dn, Fn);

  // residual + LN2 -> output
  hipLaunchKernelGGL(add_ln_kernel, dim3(Mrows / 8), dim3(256), 0, stream,
                     h1f, f2f, g2, beta2, out, (__bf16*)nullptr);
}
